// MLPNonLinearFieldChargesBlock_13872744366317
// MI455X (gfx1250) — matrix operand order repaired and checked by reading, not compiled, not run on hardware
//
#include <hip/hip_runtime.h>
#include <stdint.h>

// ---------------------------------------------------------------------------
// MLPNonLinearFieldChargesBlock for MI455X (gfx1250), wave32 + WMMA bf16.
//
// Roofline: ~41 GFLOP of GEMM work vs ~410MB mandatory HBM traffic
// (17.6us @ 23.3TB/s). bf16 WMMA (16x16x32, f32 accum) is the precision the
// chip rewards; fp32 WMMA (16x16x4) would be ~8x slower per instruction.
// All scale factors (1/sqrt(128), 1/16, 0.25, 1/sqrt(3)) are folded into the
// prepped bf16 weights. f32->bf16 conversions use v_cvt_pk_bf16_f32.
// ---------------------------------------------------------------------------

#define NNODES 100000
#define TILES  (NNODES / 16)   // 6250 exact
#define ROWF   512

typedef __attribute__((ext_vector_type(16))) __bf16 v16bf;
typedef __attribute__((ext_vector_type(2)))  __bf16 bf16x2;
typedef __attribute__((ext_vector_type(2)))  float  f32x2;
typedef __attribute__((ext_vector_type(8)))  float  v8f;

#define SILU_CST   1.6765056f     // 1/sqrt(E[silu(z)^2]), z~N(0,1)
#define INV_SQRT2  0.70710678f
#define INV_SQRT3  0.57735027f
#define UP_SCALE   0.088388348f   // 1/sqrt(128)
#define OUT_SCALE  0.0625f        // 1/16
#define W4_SCALE   0.25f

// bf16 weight pack layout in d_ws (element offsets)
#define OFF_UP0 0                 // W_up_0e^T  [v][u]   128x128
#define OFF_UP1 16384             // W_up_1o^T  [v][u]   128x128
#define OFF_WO0 32768             // W_out_0e^T [v'][u]  128x256
#define OFF_WO1 65536             // W_out_1o^T [v'][u]  128x256
#define OFF_W4  98304             // W_mlp4^T   [c][k]   512x16
#define WS_TOT  106496

__device__ __forceinline__ unsigned short f2bf(float f) {
  union { float f; unsigned u; } v; v.f = f;
  unsigned r = v.u + 0x7fffu + ((v.u >> 16) & 1u);   // round-to-nearest-even
  return (unsigned short)(r >> 16);
}

// packed f32x2 -> bf16x2 (one v_cvt_pk_bf16_f32 when available)
__device__ __forceinline__ unsigned pack2bf(float a, float b) {
  union { bf16x2 v; unsigned u; } c;
#if __has_builtin(__builtin_amdgcn_cvt_pk_bf16_f32)
  c.v = __builtin_amdgcn_cvt_pk_bf16_f32(a, b);
#else
  f32x2 f = {a, b};
  c.v = __builtin_convertvector(f, bf16x2);
#endif
  return c.u;
}

// --------------------------- weight prep kernel ----------------------------
__global__ void prep_weights(const float* __restrict__ w0e,
                             const float* __restrict__ w1o,
                             const float* __restrict__ wo0,
                             const float* __restrict__ wo1,
                             const float* __restrict__ wm4,
                             unsigned short* __restrict__ ws) {
  for (int i = blockIdx.x * blockDim.x + threadIdx.x; i < WS_TOT;
       i += gridDim.x * blockDim.x) {
    float val;
    if (i < OFF_UP1)      { int t = i;           int v = t >> 7, u = t & 127; val = w0e[u * 128 + v] * UP_SCALE; }
    else if (i < OFF_WO0) { int t = i - OFF_UP1; int v = t >> 7, u = t & 127; val = w1o[u * 128 + v] * UP_SCALE; }
    else if (i < OFF_WO1) { int t = i - OFF_WO0; int v = t >> 8, u = t & 255; val = wo0[u * 128 + v] * OUT_SCALE; }
    else if (i < OFF_W4)  { int t = i - OFF_WO1; int v = t >> 8, u = t & 255; val = wo1[u * 128 + v] * OUT_SCALE; }
    else {
      int t = i - OFF_W4; int c = t >> 4, k = t & 15;
      float s = W4_SCALE;
      if (c >= 128 && c < 256) s *= INV_SQRT3;   // fold 1/sqrt(3) into w110 gate
      val = wm4[k * 512 + c] * s;
    }
    ws[i] = f2bf(val);
  }
}

// ------------------------------ main kernel --------------------------------
union AFrag {
  v16bf v;
  unsigned short s[16];
  unsigned u[8];
  uint4 q[2];
};

#define A2_STRIDE 264   // (264*2/4) mod 64 = 4 banks/row -> conflict-free frag reads

struct WaveBuf {
  unsigned short a2[4][16 * A2_STRIDE]; // staging: A matrices of the out-GEMMs (bf16)
  unsigned short hbf[256];              // h tile (16 nodes x 16, bf16)
  float4 pbuf[16];                      // p0,p1x,p1y,p1z per node
};

__launch_bounds__(128)
__global__ void charges_kernel(const float* __restrict__ nf,
                               const float* __restrict__ pot,
                               const float* __restrict__ wm1,
                               const float* __restrict__ wm2,
                               const float* __restrict__ wm3,
                               const unsigned short* __restrict__ ws,
                               float* __restrict__ out) {
  __shared__ WaveBuf lds[4];           // 34,560 B per wave, 138,240 B total
  const int wave  = threadIdx.x >> 5;
  const int lane  = threadIdx.x & 31;
  const int lcol  = lane & 15;         // D-layout column / A-layout row
  const int lhalf = lane >> 4;
  const int rbase = lhalf * 8;         // D-layout row base
  WaveBuf& W = lds[wave];

  const unsigned short* up0T = ws + OFF_UP0;
  const unsigned short* up1T = ws + OFF_UP1;
  const unsigned short* wo0T = ws + OFF_WO0;
  const unsigned short* wo1T = ws + OFF_WO1;
  const unsigned short* w4T  = ws + OFF_W4;

  for (int tile = blockIdx.x * 4 + wave; tile < TILES; tile += gridDim.x * 4) {
    const int tb = tile * 16;

    // ---- phase 0: per-node tiny MLP -> h (bf16 in LDS), stage p ----------
    if (lane < 16) {
      int node = tb + lane;
      float4 pv = ((const float4*)pot)[node];
      W.pbuf[lane] = pv;
      float i0 = fabsf(pv.x) + 1.0f;
      float i1 = sqrtf(pv.y * pv.y + pv.z * pv.z + pv.w * pv.w) + 1.0f;
      float h1[16], h2[16], h3[16];
#pragma unroll
      for (int k = 0; k < 16; k++) {
        float t = (i0 * wm1[k] + i1 * wm1[16 + k]) * INV_SQRT2;
        h1[k] = SILU_CST * t / (1.0f + __expf(-t));
      }
#pragma unroll
      for (int k = 0; k < 16; k++) {
        float acc = 0.f;
#pragma unroll
        for (int j2 = 0; j2 < 16; j2++) acc += h1[j2] * wm2[j2 * 16 + k];
        float t = acc * 0.25f;
        h2[k] = SILU_CST * t / (1.0f + __expf(-t));
      }
#pragma unroll
      for (int k = 0; k < 16; k++) {
        float acc = 0.f;
#pragma unroll
        for (int j2 = 0; j2 < 16; j2++) acc += h2[j2] * wm3[j2 * 16 + k];
        float t = acc * 0.25f;
        h3[k] = SILU_CST * t / (1.0f + __expf(-t));
      }
      unsigned* hw = (unsigned*)&W.hbf[lane * 16];
#pragma unroll
      for (int k = 0; k < 8; k++) hw[k] = pack2bf(h3[2 * k], h3[2 * k + 1]);
    }
    // DS ops are in-order within a wave -> no barrier needed (per-wave buffers).

    // h A-fragment (K padded 16->32 with zeros), j-invariant
    AFrag ha;
    ha.q[0] = lhalf ? make_uint4(0, 0, 0, 0) : *(const uint4*)&W.hbf[lcol * 16];
    ha.q[1] = lhalf ? make_uint4(0, 0, 0, 0) : *(const uint4*)&W.hbf[lcol * 16 + 8];

    // ---- build A fragments for the up-GEMMs (x0, x1[m]) -------------------
    // 16-bit A layout: lane holds row M=lcol; K = kb*32 + (lhalf?8:0) + {0..7, 16..23}
    const float* xr = nf + (size_t)(tb + lcol) * ROWF;
    AFrag a0[4], a1[3][4];
#pragma unroll
    for (int kb = 0; kb < 4; kb++) {
      int kbase = kb * 32 + rbase;
#pragma unroll
      for (int i = 0; i < 4; i++) {
        a0[kb].u[i]     = pack2bf(xr[kbase + 2 * i],      xr[kbase + 2 * i + 1]);
        a0[kb].u[4 + i] = pack2bf(xr[kbase + 16 + 2 * i], xr[kbase + 17 + 2 * i]);
      }
      const float* x1r = xr + 128;   // x1[u][m] at column 128 + u*3 + m
#pragma unroll
      for (int i = 0; i < 4; i++) {
        int ua = kbase + 2 * i, ub = kbase + 2 * i + 1;
        int uc = ua + 16,       ud = ub + 16;
#pragma unroll
        for (int m = 0; m < 3; m++) {
          a1[m][kb].u[i]     = pack2bf(x1r[ua * 3 + m], x1r[ub * 3 + m]);
          a1[m][kb].u[4 + i] = pack2bf(x1r[uc * 3 + m], x1r[ud * 3 + m]);
        }
      }
    }

    // prefetch next tile's node_feats row (global_prefetch_b8)
    {
      int ntile = tile + gridDim.x * 4;
      if (ntile < TILES) {
        const float* nxr = nf + (size_t)(ntile * 16 + lcol) * ROWF + lhalf * 256;
#pragma unroll
        for (int q = 0; q < 8; q++) __builtin_prefetch(nxr + q * 32, 0, 1);
      }
    }

    // ---- phase 1: up GEMMs + gate WMMA + equivariant elementwise ----------
#pragma unroll 1
    for (int j = 0; j < 8; j++) {
      v8f y0 = {};
      v8f y1[3] = {{}, {}, {}};
#pragma unroll
      for (int kb = 0; kb < 4; kb++) {
        // B layout: lane holds column v=j*16+lcol; K = kb*32 + lhalf*8 + {0..7, 16..23}
        AFrag b0, b1;
        const unsigned short* p0w = up0T + (j * 16 + lcol) * 128 + kb * 32 + lhalf * 8;
        b0.q[0] = *(const uint4*)p0w;
        b0.q[1] = *(const uint4*)(p0w + 16);
        y0 = __builtin_amdgcn_wmma_f32_16x16x32_bf16(false, a0[kb].v, false, b0.v,
                                                     (short)0, y0, false, false);
        const unsigned short* p1w = up1T + (j * 16 + lcol) * 128 + kb * 32 + lhalf * 8;
        b1.q[0] = *(const uint4*)p1w;
        b1.q[1] = *(const uint4*)(p1w + 16);
#pragma unroll
        for (int m = 0; m < 3; m++)
          y1[m] = __builtin_amdgcn_wmma_f32_16x16x32_bf16(false, a1[m][kb].v, false, b1.v,
                                                          (short)0, y1[m], false, false);
      }

      // gate tiles w00/w110/w011/w101 = h @ W_mlp4 (K padded to 32)
      v8f wt[4];
#pragma unroll
      for (int blk = 0; blk < 4; blk++) {
        AFrag bw;
        if (lhalf == 0) {
          const unsigned short* pw = w4T + (blk * 128 + j * 16 + lcol) * 16;
          bw.q[0] = *(const uint4*)pw;
          bw.q[1] = *(const uint4*)(pw + 8);
        } else {
          bw.q[0] = make_uint4(0, 0, 0, 0);  // K=16..31 is zero padding
          bw.q[1] = make_uint4(0, 0, 0, 0);
        }
        v8f z = {};
        wt[blk] = __builtin_amdgcn_wmma_f32_16x16x32_bf16(false, ha.v, false, bw.v,
                                                          (short)0, z, false, false);
      }

      // elementwise in matching D-layout; stage bf16 over row pairs
#pragma unroll
      for (int rp = 0; rp < 8; rp += 2) {
        float oa[2][8];
#pragma unroll
        for (int rr = 0; rr < 2; rr++) {
          int r = rp + rr;
          float4 p = W.pbuf[rbase + r];
          float y0v = y0[r];
          float d  = y1[0][r] * p.y + y1[1][r] * p.z + y1[2][r] * p.w;
          float w00p0 = wt[0][r] * p.x;
          float w011y = wt[2][r] * y0v;
          float w101p = wt[3][r] * p.x;
          oa[rr][0] = w00p0 * y0v;           // out0a
          oa[rr][1] = wt[1][r] * d;          // out0b (1/sqrt3 folded in weights)
          oa[rr][2] = w011y * p.y;           // out1a m=0
          oa[rr][3] = w101p * y1[0][r];      // out1b m=0
          oa[rr][4] = w011y * p.z;
          oa[rr][5] = w101p * y1[1][r];
          oa[rr][6] = w011y * p.w;
          oa[rr][7] = w101p * y1[2][r];
        }
        int row0 = (rbase + rp) * A2_STRIDE + j * 16 + lcol;
        int row1 = row0 + A2_STRIDE;
#pragma unroll
        for (int slot = 0; slot < 8; slot++) {
          unsigned pk = pack2bf(oa[0][slot], oa[1][slot]);
          int mat = slot >> 1;
          int off = (slot & 1) * 128;
          W.a2[mat][row0 + off] = (unsigned short)(pk & 0xffffu);
          W.a2[mat][row1 + off] = (unsigned short)(pk >> 16);
        }
      }
    } // j

    // ---- phase 2: out GEMMs (K=256), write charges ------------------------
#pragma unroll 1
    for (int mat = 0; mat < 4; mat++) {
      AFrag af[8];
#pragma unroll
      for (int kb = 0; kb < 8; kb++) {
        const unsigned short* pa = &W.a2[mat][lcol * A2_STRIDE + kb * 32 + lhalf * 16];
        af[kb].q[0] = *(const uint4*)pa;
        af[kb].q[1] = *(const uint4*)(pa + 8);
      }
      const unsigned short* wT = (mat == 0) ? wo0T : wo1T;
#pragma unroll 1
      for (int jj = 0; jj < 8; jj++) {
        v8f acc = {};
#pragma unroll
        for (int kb = 0; kb < 8; kb++) {
          AFrag b;
          const unsigned short* pb = wT + (jj * 16 + lcol) * 256 + kb * 32 + lhalf * 16;
          b.q[0] = *(const uint4*)pb;
          b.q[1] = *(const uint4*)(pb + 8);
          acc = __builtin_amdgcn_wmma_f32_16x16x32_bf16(false, af[kb].v, false, b.v,
                                                        (short)0, acc, false, false);
        }
#pragma unroll
        for (int r = 0; r < 8; r++) {
          int node = tb + rbase + r;
          int col = (mat == 0) ? (jj * 16 + lcol)
                               : (128 + (jj * 16 + lcol) * 3 + (mat - 1));
          out[(size_t)node * ROWF + col] = acc[r];
        }
      }
    }
  } // tile loop
}

// ------------------------------- launcher ----------------------------------
extern "C" void kernel_launch(void* const* d_in, const int* in_sizes, int n_in,
                              void* d_out, int out_size, void* d_ws, size_t ws_size,
                              hipStream_t stream) {
  const float* nf  = (const float*)d_in[0];  // node_feats (N,512)
  const float* pot = (const float*)d_in[1];  // pot_feats  (N,4)
  const float* w0e = (const float*)d_in[2];  // W_up_0e
  const float* w1o = (const float*)d_in[3];  // W_up_1o
  const float* wm1 = (const float*)d_in[4];  // W_mlp1 (2,16)
  const float* wm2 = (const float*)d_in[5];  // W_mlp2 (16,16)
  const float* wm3 = (const float*)d_in[6];  // W_mlp3 (16,16)
  const float* wm4 = (const float*)d_in[7];  // W_mlp4 (16,512)
  const float* wo0 = (const float*)d_in[8];  // W_out_0e (256,128)
  const float* wo1 = (const float*)d_in[9];  // W_out_1o (256,128)
  float* out = (float*)d_out;
  unsigned short* ws = (unsigned short*)d_ws;

  prep_weights<<<(WS_TOT + 255) / 256, 256, 0, stream>>>(w0e, w1o, wo0, wo1, wm4, ws);
  charges_kernel<<<784, 128, 0, stream>>>(nf, pot, wm1, wm2, wm3, ws, out);
}
